// Glm4MoeTopKRouter_3246995276391
// MI455X (gfx1250) — matrix operand order repaired and checked by reading, not compiled, not run on hardware
//
#include <hip/hip_runtime.h>
#include <hip/hip_bf16.h>
#include <stdint.h>

// ---------------------------------------------------------------------------
// GLM4 MoE TopK router for MI455X (gfx1250, wave32, WMMA).
//   hs   : [T=16384, H=4096] bf16
//   W    : [H=4096, E=128]   bf16
//   bias : [E=128] f32
//   out  : [T, 8] f32
// HBM-bound on hs (134 MB @ 23.3 TB/s ~ 5.8us). GEMM via v_wmma_f32_16x16x32_bf16.
// Each wave: 4 M-tiles (64 tokens) x 8 N-tiles (128 experts), K-loop of 128x32.
// B pre-repacked into WMMA fragment layout (1 MB, lives in L2). A/B fragment
// sets are double-buffered so loads of step s+1 overlap the 32-WMMA burst of
// step s (graduated s_wait_loadcnt instead of wait-0).
// ---------------------------------------------------------------------------

typedef __attribute__((ext_vector_type(16))) __bf16       v16bf;
typedef __attribute__((ext_vector_type(8)))  float        v8f;
typedef __attribute__((ext_vector_type(4)))  unsigned int u32x4;

union Frag { u32x4 q[2]; v16bf v; };

#define H_DIM   4096
#define E_DIM   128
#define KSTEPS  (H_DIM / 32)       // 128 k-steps of 32
#define M_TILES 4                  // 16-token WMMA tiles per wave
#define TOK_PER_BLK (M_TILES * 16) // 64 tokens, one wave per block
#define TOKPAD  65                 // LDS token stride (bank-conflict-free)
#define ROW_U4  (16 * H_DIM / 8)   // u32x4 elems per 16-row M-tile stride (8192)

// --- Repack W into WMMA B-fragment order -----------------------------------
// B frag (32x16 bf16) per ISA 7.12.2: lane<16 -> col n=lane, K=0..7 then
// K16..23 packed 2/dword over 8 dwords; lane>=16 -> col n=lane-16, K8..15 then K24..31.
// Packed dword index = ((step*8 + ntile)*32 + lane)*8 + j
__global__ void repack_b_kernel(const __hip_bfloat16* __restrict__ kern,
                                uint32_t* __restrict__ bp) {
  int i    = blockIdx.x * blockDim.x + threadIdx.x;   // 0 .. 262143
  int j    = i & 7;
  int lane = (i >> 3) & 31;
  int nt   = (i >> 8) & 7;
  int step = i >> 11;
  int n    = nt * 16 + (lane & 15);
  int k0   = step * 32 + ((lane & 16) ? 8 : 0) + 2 * (j & 3) + ((j & 4) ? 16 : 0);
  const uint16_t* kb = (const uint16_t*)kern;
  uint32_t lo = kb[(size_t)k0 * E_DIM + n];
  uint32_t hi = kb[(size_t)(k0 + 1) * E_DIM + n];
  bp[i] = lo | (hi << 16);
}

// --- fragment load / math helpers ------------------------------------------
__device__ __forceinline__ void load_a(Frag (&a)[M_TILES],
                                       const u32x4* __restrict__ aBase, int s) {
#pragma unroll
  for (int t = 0; t < M_TILES; ++t) {
    a[t].q[0] = aBase[(size_t)s * 4 + t * ROW_U4];
    a[t].q[1] = aBase[(size_t)s * 4 + t * ROW_U4 + 2];
  }
}

__device__ __forceinline__ void load_b(Frag (&b)[8],
                                       const u32x4* __restrict__ bq, int s) {
#pragma unroll
  for (int nt = 0; nt < 8; ++nt) {
    size_t fi = ((size_t)s * 8 + nt) * 64;    // u32x4 units
    b[nt].q[0] = bq[fi];
    b[nt].q[1] = bq[fi + 1];
  }
}

__device__ __forceinline__ void wmma_block(v8f (&acc)[M_TILES][8],
                                           const Frag (&a)[M_TILES],
                                           const Frag (&b)[8]) {
#pragma unroll
  for (int nt = 0; nt < 8; ++nt)
#pragma unroll
    for (int t = 0; t < M_TILES; ++t)
      acc[t][nt] = __builtin_amdgcn_wmma_f32_16x16x32_bf16(
          false, a[t].v, false, b[nt].v, (short)0, acc[t][nt], false, false);
}

// --- GEMM + grouped top-k routing ------------------------------------------
__global__ void __launch_bounds__(32, 1)
moe_router_kernel(const __hip_bfloat16* __restrict__ hs,
                  const uint32_t* __restrict__ bpack,
                  const float* __restrict__ bias,
                  float* __restrict__ out) {
  __shared__ float sc[E_DIM][TOKPAD];           // [expert][token-in-block]

  const int lane = threadIdx.x & 31;
  const int col  = lane & 15;
  const int hi16 = (lane >> 4) & 1;

  const size_t blockTok = (size_t)blockIdx.x * TOK_PER_BLK;

  // A fragment addressing (16-bit A 16x32, ISA 7.12.2):
  // lane<16: row M=lane, K0..7 then K16..23; lane>=16: row M=lane-16,
  // K8..15 then K24..31 -> two b128 loads at elem (koff) and (koff+16).
  const size_t gA = (blockTok + col) * (size_t)H_DIM + (hi16 ? 8 : 0);
  const u32x4* aBase = (const u32x4*)((const uint16_t*)hs + gA);
  const u32x4* bq    = (const u32x4*)bpack + (size_t)lane * 2;

  v8f zero = {0.f, 0.f, 0.f, 0.f, 0.f, 0.f, 0.f, 0.f};
  v8f acc[M_TILES][8];
#pragma unroll
  for (int t = 0; t < M_TILES; ++t)
#pragma unroll
    for (int nt = 0; nt < 8; ++nt) acc[t][nt] = zero;

  // ---- software-pipelined K loop: double-buffered fragment sets ----
  Frag a0[M_TILES], a1[M_TILES], b0[8], b1[8];
  load_a(a0, aBase, 0);
  load_b(b0, bq, 0);

#pragma unroll 1
  for (int s = 0; s < KSTEPS - 2; s += 2) {
    load_a(a1, aBase, s + 1);
    load_b(b1, bq, s + 1);
    if (s + 10 < KSTEPS) {                     // A-stream prefetch, WGP scope
#pragma unroll
      for (int t = 0; t < M_TILES; ++t) {
        __builtin_prefetch((const void*)(aBase + (s + 8) * 4 + t * ROW_U4), 0, 3);
        __builtin_prefetch((const void*)(aBase + (s + 9) * 4 + t * ROW_U4), 0, 3);
      }
    }
    wmma_block(acc, a0, b0);                   // covers latency of (s+1) loads
    load_a(a0, aBase, s + 2);
    load_b(b0, bq, s + 2);
    wmma_block(acc, a1, b1);                   // covers latency of (s+2) loads
  }
  // tail: s = KSTEPS-2, KSTEPS-1
  load_a(a1, aBase, KSTEPS - 1);
  load_b(b1, bq, KSTEPS - 1);
  wmma_block(acc, a0, b0);
  wmma_block(acc, a1, b1);

  // ---- spill logits to LDS: C/D layout: lane<16 rows v(0..7), lane>=16
  // rows v+8; column = lane&15. Same-wave DS is in-order: no barrier. ----
#pragma unroll
  for (int t = 0; t < M_TILES; ++t) {
    const int tokb = t * 16 + hi16 * 8;
#pragma unroll
    for (int nt = 0; nt < 8; ++nt)
#pragma unroll
      for (int v = 0; v < 8; ++v)
        sc[nt * 16 + col][tokb + v] = acc[t][nt][v];
  }

  // ---- per-token routing: lane handles tokens lane and lane+32 ----
#pragma unroll 1
  for (int tt = 0; tt < 2; ++tt) {
    const int tok = tt * 32 + lane;

    // pass 1: sigmoid + bias, per-group top-2 sums (unrolled -> registers)
    float gsum[8];
#pragma unroll
    for (int g = 0; g < 8; ++g) {
      float m1 = -1e30f, m2 = -1e30f;
#pragma unroll
      for (int j = 0; j < 16; ++j) {
        int e = g * 16 + j;
        float x = sc[e][tok];
        float s = 1.0f / (1.0f + __expf(-x));
        float v = s + bias[e];                  // uniform addr -> scalar loads
        sc[e][tok] = v;
        if (v > m1) { m2 = m1; m1 = v; }
        else if (v > m2) { m2 = v; }
      }
      gsum[g] = m1 + m2;
    }

    // top-4 groups by rank (ties -> lower index, matching jax top_k)
    bool keep[8];
#pragma unroll
    for (int g = 0; g < 8; ++g) {
      int rank = 0;
#pragma unroll
      for (int h = 0; h < 8; ++h)
        rank += (gsum[h] > gsum[g]) || ((h < g) && (gsum[h] == gsum[g]));
      keep[g] = (rank < 4);
    }

    // mask non-selected groups to 0.0 (as in reference)
#pragma unroll
    for (int e = 0; e < E_DIM; ++e) {
      float v = sc[e][tok];
      sc[e][tok] = keep[e >> 4] ? v : 0.0f;
    }

    // 8 argmax-and-remove passes; recover sigmoid weight = sc - bias[idx]
    float denom = 1e-20f;
    float w0, w1, w2, w3, w4, w5, w6, w7;
#pragma unroll
    for (int k = 0; k < 8; ++k) {
      float best = -1e30f; int bi = 0;
      for (int e = 0; e < E_DIM; ++e) {
        float v = sc[e][tok];
        if (v > best) { best = v; bi = e; }
      }
      sc[bi][tok] = -1e30f;
      float wk = best - bias[bi];
      denom += wk;
      switch (k) {
        case 0: w0 = wk; break; case 1: w1 = wk; break;
        case 2: w2 = wk; break; case 3: w3 = wk; break;
        case 4: w4 = wk; break; case 5: w5 = wk; break;
        case 6: w6 = wk; break; default: w7 = wk; break;
      }
    }

    const float scale = 2.5f / denom;
    float* o = out + (blockTok + (size_t)tok) * 8;
    o[0] = w0 * scale; o[1] = w1 * scale; o[2] = w2 * scale; o[3] = w3 * scale;
    o[4] = w4 * scale; o[5] = w5 * scale; o[6] = w6 * scale; o[7] = w7 * scale;
  }
}

// ---------------------------------------------------------------------------
extern "C" void kernel_launch(void* const* d_in, const int* in_sizes, int n_in,
                              void* d_out, int out_size, void* d_ws, size_t ws_size,
                              hipStream_t stream) {
  const __hip_bfloat16* hs   = (const __hip_bfloat16*)d_in[0];
  const __hip_bfloat16* kern = (const __hip_bfloat16*)d_in[1];
  const float*          bias = (const float*)d_in[2];
  float*    out   = (float*)d_out;
  uint32_t* bpack = (uint32_t*)d_ws;          // 1 MB packed B

  const int T = in_sizes[0] / H_DIM;          // 16384 tokens

  const int packElems = KSTEPS * 8 * 32 * 8;  // 262144 dwords
  repack_b_kernel<<<packElems / 256, 256, 0, stream>>>(kern, bpack);

  moe_router_kernel<<<T / TOK_PER_BLK, 32, 0, stream>>>(hs, bpack, bias, out);
}
